// ConvFrameMaskDecoder_15324443312411
// MI455X (gfx1250) — matrix operand order repaired and checked by reading, not compiled, not run on hardware
//
#include <hip/hip_runtime.h>
#include <hip/hip_bf16.h>

// ---------------- constants ----------------
#define B_      8
#define T_      24
#define L_      145
#define DHID    1024
#define DFRAME  2500
#define DEMB    128
#define DGCN    512
#define V_      1024
#define DINP    4164      // DHID+DFRAME+DEMB+DGCN... actually DFRAME+DHID+DEMB+DGCN
#define DCONT   5188
#define FLAT    3136
#define K32INP  4192      // DINP padded to 32
#define K32CONT 5216      // DCONT padded to 32
#define BNI     0.9999950003749064f   // 1/sqrt(1+1e-5)

typedef __attribute__((ext_vector_type(16))) __bf16 v16bf;
typedef __attribute__((ext_vector_type(8)))  float  v8f;

union Frag { uint4 q[2]; v16bf v; };

__device__ __forceinline__ unsigned short f2bf(float f) {
  unsigned int u = __float_as_uint(f);
  u += 0x7FFFu + ((u >> 16) & 1u);            // round-to-nearest-even
  return (unsigned short)(u >> 16);
}
__device__ __forceinline__ float sigm_(float x) { return 1.f / (1.f + __expf(-x)); }

// ---------------- WMMA GEMM ----------------
// C[M,N] (ldc) = act( bn( A[Mp,K32]bf16 * B[Np,K32]bf16^T + bias ) ), optional +=C
// mode: 0 = bias, 1 = bias+relu, 2 = bias -> BN(eval) -> relu
__global__ __launch_bounds__(256) void gemm_bf16_wmma(
    const unsigned short* __restrict__ A, const unsigned short* __restrict__ Bw,
    float* __restrict__ C, int ldc,
    const float* __restrict__ bias, const float* __restrict__ g, const float* __restrict__ bb,
    int M, int N, int K32, int mode, int accum)
{
  const int lane = threadIdx.x & 31;
  const int wave = (blockIdx.x << 3) + (threadIdx.x >> 5);
  const int tilesN = (N + 15) >> 4;
  const int tilesM = (M + 15) >> 4;
  if (wave >= tilesM * tilesN) return;           // wave-uniform exit
  const int m0 = (wave / tilesN) << 4;
  const int n0 = (wave % tilesN) << 4;
  const int l16 = lane & 15;
  const int lh  = lane >> 4;

  // A fragment: lane holds row m0+l16; K-halves per ISA 16-bit A layout
  const unsigned short* Ap = A + (size_t)(m0 + l16) * K32 + lh * 8;
  // B fragment: lane holds column n0+l16 (weight row, padded to Np); K-halves same as A
  const unsigned short* Bp = Bw + (size_t)(n0 + l16) * K32 + lh * 8;

  v8f acc = {0.f, 0.f, 0.f, 0.f, 0.f, 0.f, 0.f, 0.f};
  for (int k = 0; k < K32; k += 32) {
    Frag a, b;
    a.q[0] = *reinterpret_cast<const uint4*>(Ap + k);
    a.q[1] = *reinterpret_cast<const uint4*>(Ap + k + 16);
    b.q[0] = *reinterpret_cast<const uint4*>(Bp + k);
    b.q[1] = *reinterpret_cast<const uint4*>(Bp + k + 16);
    acc = __builtin_amdgcn_wmma_f32_16x16x32_bf16(false, a.v, false, b.v,
                                                  (short)0, acc, false, false);
  }

  const int n = n0 + l16;
  if (n < N) {
    #pragma unroll
    for (int r = 0; r < 8; ++r) {
      int m = m0 + r + lh * 8;                  // C layout: VGPR r -> M=r(+8), N=lane
      if (m < M) {
        float x = acc[r];
        if (accum) x += C[(size_t)m * ldc + n];
        if (bias)  x += bias[n];
        if (mode == 2) x = g[n] * (x * BNI) + bb[n];
        if (mode >= 1) x = fmaxf(x, 0.f);
        C[(size_t)m * ldc + n] = x;
      }
    }
  }
}

// ---------------- pack kernels (fp32 -> padded bf16) ----------------
__global__ void pack_weight(const float* __restrict__ W, unsigned short* __restrict__ out,
                            int N, int K, int K32, int Np) {
  size_t i = (size_t)blockIdx.x * 256 + threadIdx.x;
  if (i >= (size_t)Np * K32) return;
  int n = (int)(i / K32), k = (int)(i % K32);
  out[i] = (n < N && k < K) ? f2bf(W[(size_t)n * K + k]) : (unsigned short)0;
}

__global__ void pack_act(const float* __restrict__ X, unsigned short* __restrict__ out,
                         int M, int K, int Mp, int K32) {
  size_t i = (size_t)blockIdx.x * 256 + threadIdx.x;
  if (i >= (size_t)Mp * K32) return;
  int m = (int)(i / K32), k = (int)(i % K32);
  out[i] = (m < M && k < K) ? f2bf(X[(size_t)m * K + k]) : (unsigned short)0;
}

__global__ void pack_frames(const float* __restrict__ frames, unsigned short* __restrict__ out, int t) {
  int i = blockIdx.x * 256 + threadIdx.x;          // out [400,512]
  if (i >= 400 * 512) return;
  int m = i >> 9, k = i & 511;
  if (m < 392) {
    int b = m / 49, hw = m % 49;
    out[i] = f2bf(frames[(((size_t)b * T_ + t) * 512 + k) * 49 + hw]);
  } else out[i] = 0;
}

__global__ void pack_fc(const float* __restrict__ C2, unsigned short* __restrict__ out) {
  int i = blockIdx.x * 256 + threadIdx.x;          // out [16,3136], k = o*49+hw
  if (i >= 16 * FLAT) return;
  int m = i / FLAT, k = i % FLAT;
  if (m < B_) {
    int o = k / 49, hw = k % 49;
    out[i] = f2bf(C2[((size_t)m * 49 + hw) * 64 + o]);
  } else out[i] = 0;
}

__global__ void pack_inp(const float* __restrict__ vis, const float* __restrict__ wgt,
                         const float* __restrict__ e, const float* __restrict__ gcn,
                         unsigned short* __restrict__ out) {
  int i = blockIdx.x * 256 + threadIdx.x;          // out [16,K32INP]
  if (i >= 16 * K32INP) return;
  int m = i / K32INP, k = i % K32INP;
  float v = 0.f; bool ok = (m < B_);
  if (ok) {
    if      (k < 2500) v = vis[m * DFRAME + k];
    else if (k < 3524) v = wgt[m * DHID + (k - 2500)];
    else if (k < 3652) v = e  [m * DEMB + (k - 3524)];
    else if (k < 4164) v = gcn[m * DGCN + (k - 3652)];
    else ok = false;
  }
  out[i] = ok ? f2bf(v) : (unsigned short)0;
}

__global__ void pack_cont(const float* __restrict__ h, const float* __restrict__ vis,
                          const float* __restrict__ wgt, const float* __restrict__ e,
                          const float* __restrict__ gcn, unsigned short* __restrict__ out) {
  int i = blockIdx.x * 256 + threadIdx.x;          // out [16,K32CONT]
  if (i >= 16 * K32CONT) return;
  int m = i / K32CONT, k = i % K32CONT;
  float v = 0.f; bool ok = (m < B_);
  if (ok) {
    if      (k < 1024) v = h  [m * DHID + k];
    else if (k < 3524) v = vis[m * DFRAME + (k - 1024)];
    else if (k < 4548) v = wgt[m * DHID + (k - 3524)];
    else if (k < 4676) v = e  [m * DEMB + (k - 4548)];
    else if (k < 5188) v = gcn[m * DGCN + (k - 4676)];
    else ok = false;
  }
  out[i] = ok ? f2bf(v) : (unsigned short)0;
}

// ---------------- attention ----------------
__global__ void attn_logits(const float* __restrict__ enc, const float* __restrict__ hq,
                            float* __restrict__ logits) {
  int w = ((blockIdx.x * blockDim.x) + threadIdx.x) >> 5;
  int lane = threadIdx.x & 31;
  if (w >= B_ * L_) return;
  int b = w / L_, l = w % L_;
  const float* er = enc + ((size_t)b * L_ + l) * DHID;
  const float* hr = hq + (size_t)b * DHID;
  float s = 0.f;
  for (int d = lane; d < DHID; d += 32) s += er[d] * hr[d];
  #pragma unroll
  for (int off = 16; off > 0; off >>= 1) s += __shfl_xor(s, off, 32);
  if (lane == 0) logits[b * L_ + l] = s;
}

__global__ void softmax_attn(const float* __restrict__ logits, float* __restrict__ score,
                             float* __restrict__ attn_out, int t) {
  int b = threadIdx.x >> 5, lane = threadIdx.x & 31;
  if (b >= B_) return;
  const float* lg = logits + b * L_;
  float mx = -1e30f;
  for (int l = lane; l < L_; l += 32) mx = fmaxf(mx, lg[l]);
  #pragma unroll
  for (int off = 16; off > 0; off >>= 1) mx = fmaxf(mx, __shfl_xor(mx, off, 32));
  float sum = 0.f;
  for (int l = lane; l < L_; l += 32) sum += __expf(lg[l] - mx);
  #pragma unroll
  for (int off = 16; off > 0; off >>= 1) sum += __shfl_xor(sum, off, 32);
  float inv = 1.f / sum;
  for (int l = lane; l < L_; l += 32) {
    float v = __expf(lg[l] - mx) * inv;
    score[b * L_ + l] = v;
    attn_out[((size_t)b * T_ + t) * L_ + l] = v;
  }
}

__global__ void attn_weighted(const float* __restrict__ enc, const float* __restrict__ score,
                              float* __restrict__ wgt) {
  int i = blockIdx.x * 256 + threadIdx.x;
  if (i >= B_ * DHID) return;
  int b = i >> 10, d = i & 1023;
  float s = 0.f;
  for (int l = 0; l < L_; ++l) s += score[b * L_ + l] * enc[((size_t)b * L_ + l) * DHID + d];
  wgt[i] = s;
}

// ---------------- LSTM pointwise ----------------
__global__ void lstm_pw(const float* __restrict__ gates, float* __restrict__ h, float* __restrict__ c) {
  int i = blockIdx.x * 256 + threadIdx.x;
  if (i >= B_ * DHID) return;
  int b = i >> 10, j = i & 1023;
  const float* gr = gates + (size_t)b * 4 * DHID;
  float ig = sigm_(gr[j]), fg = sigm_(gr[DHID + j]);
  float gg = tanhf(gr[2 * DHID + j]), og = sigm_(gr[3 * DHID + j]);
  float cn = fg * c[i] + ig * gg;
  c[i] = cn;
  h[i] = og * tanhf(cn);
}

// ---------------- argmax + embedding gather ----------------
__global__ void argmax_embed(const float* __restrict__ action, int ldc,
                             const float* __restrict__ emb, float* __restrict__ e) {
  int b = threadIdx.x >> 5, lane = threadIdx.x & 31;
  if (b >= B_) return;
  const float* ar = action + (size_t)b * ldc;
  float bv = -1e30f; int bi = 0x7FFFFFFF;
  for (int v = lane; v < V_; v += 32) {
    float x = ar[v];
    if (x > bv || (x == bv && v < bi)) { bv = x; bi = v; }
  }
  #pragma unroll
  for (int off = 16; off > 0; off >>= 1) {
    float ov = __shfl_xor(bv, off, 32);
    int   oi = __shfl_xor(bi, off, 32);
    if (ov > bv || (ov == bv && oi < bi)) { bv = ov; bi = oi; }
  }
  bi = __shfl(bi, 0, 32);
  for (int d = lane; d < DEMB; d += 32) e[b * DEMB + d] = emb[(size_t)bi * DEMB + d];
}

// ---------------- decoder kernels ----------------
__global__ void up2_nearest(const float* __restrict__ x, float* __restrict__ y, int C, int H) {
  int H2 = 2 * H;
  size_t tot = (size_t)B_ * C * H2 * H2;
  size_t i = (size_t)blockIdx.x * 256 + threadIdx.x;
  if (i >= tot) return;
  int j = (int)(i % H2); size_t r = i / H2;
  int ii = (int)(r % H2); size_t bc = r / H2;
  y[i] = x[(bc * H + (ii >> 1)) * H + (j >> 1)];
}

// ConvTranspose2d(k=4, s=2, p=1), torch weight layout [Cin, Cout, 4, 4]
__global__ void deconv4s2(const float* __restrict__ x, const float* __restrict__ w,
                          const float* __restrict__ bias, const float* __restrict__ g,
                          const float* __restrict__ bb, float* __restrict__ y,
                          int Cin, int Cout, int Hin, int mode) {
  int Ho = 2 * Hin;
  size_t tot = (size_t)B_ * Cout * Ho * Ho;
  size_t idx = (size_t)blockIdx.x * 256 + threadIdx.x;
  if (idx >= tot) return;
  int j = (int)(idx % Ho); size_t r1 = idx / Ho;
  int i = (int)(r1 % Ho); size_t r2 = r1 / Ho;
  int o = (int)(r2 % Cout); int b = (int)(r2 / Cout);
  int u0 = (i & 1) ? 0 : 1;
  int v0 = (j & 1) ? 0 : 1;
  float acc = bias[o];
  #pragma unroll
  for (int du = 0; du < 2; ++du) {
    int u = u0 + 2 * du;
    int a = (i + 1 - u) >> 1;
    if (a < 0 || a >= Hin) continue;
    #pragma unroll
    for (int dv = 0; dv < 2; ++dv) {
      int v = v0 + 2 * dv;
      int bc = (j + 1 - v) >> 1;
      if (bc < 0 || bc >= Hin) continue;
      const float* xr = x + (((size_t)b * Cin) * Hin + a) * Hin + bc;
      const float* wr = w + (size_t)o * 16 + (u * 4 + v);
      float s = 0.f;
      for (int cc = 0; cc < Cin; ++cc)
        s += xr[(size_t)cc * Hin * Hin] * wr[(size_t)cc * Cout * 16];
      acc += s;
    }
  }
  if (mode == 2) acc = g[o] * (acc * BNI) + bb[o];
  if (mode >= 1) acc = fmaxf(acc, 0.f);
  y[idx] = acc;
}

__global__ void resize_bilinear(const float* __restrict__ x, float* __restrict__ om, int t) {
  size_t idx = (size_t)blockIdx.x * 256 + threadIdx.x;
  if (idx >= (size_t)B_ * 300 * 300) return;
  int b = (int)(idx / 90000);
  int rem = (int)(idx % 90000);
  int i = rem / 300, j = rem % 300;
  const float sc = 224.f / 300.f;
  float sy = (i + 0.5f) * sc - 0.5f;
  float sx = (j + 0.5f) * sc - 0.5f;
  float y0f = floorf(sy), x0f = floorf(sx);
  float fy = sy - y0f, fx = sx - x0f;
  int y0 = (int)y0f, x0 = (int)x0f;
  int y1 = y0 + 1, x1 = x0 + 1;
  y0 = min(max(y0, 0), 223); y1 = min(max(y1, 0), 223);
  x0 = min(max(x0, 0), 223); x1 = min(max(x1, 0), 223);
  const float* xb = x + (size_t)b * 224 * 224;
  float v00 = xb[y0 * 224 + x0], v01 = xb[y0 * 224 + x1];
  float v10 = xb[y1 * 224 + x0], v11 = xb[y1 * 224 + x1];
  float v = v00 * (1.f - fy) * (1.f - fx) + v01 * (1.f - fy) * fx
          + v10 * fy * (1.f - fx) + v11 * fy * fx;
  om[((size_t)b * T_ + t) * 90000 + rem] = v;
}

// ---------------- state init / final copy ----------------
__global__ void init_state(const float* __restrict__ h0, const float* __restrict__ c0,
                           const float* __restrict__ go, float* __restrict__ h,
                           float* __restrict__ c, float* __restrict__ e) {
  int i = blockIdx.x * 256 + threadIdx.x;
  if (i < B_ * DHID) { h[i] = h0[i]; c[i] = c0[i]; }
  if (i < B_ * DEMB) e[i] = go[i & (DEMB - 1)];
}

__global__ void copy_hc(const float* __restrict__ h, const float* __restrict__ c,
                        float* __restrict__ oh, float* __restrict__ oc) {
  int i = blockIdx.x * 256 + threadIdx.x;
  if (i < B_ * DHID) { oh[i] = h[i]; oc[i] = c[i]; }
}

// ---------------- host ----------------
static inline unsigned gridFor(size_t n) { return (unsigned)((n + 255) / 256); }

extern "C" void kernel_launch(void* const* d_in, const int* in_sizes, int n_in,
                              void* d_out, int out_size, void* d_ws, size_t ws_size,
                              hipStream_t stream) {
  (void)in_sizes; (void)n_in; (void)out_size; (void)ws_size;
  const float* enc     = (const float*)d_in[0];
  const float* frames  = (const float*)d_in[1];
  const float* gcn     = (const float*)d_in[2];
  const float* h0      = (const float*)d_in[3];
  const float* c0      = (const float*)d_in[4];
  const float* conv1_w = (const float*)d_in[5];
  const float* conv1_b = (const float*)d_in[6];
  const float* bn1_g   = (const float*)d_in[7];
  const float* bn1_b   = (const float*)d_in[8];
  const float* conv2_w = (const float*)d_in[9];
  const float* conv2_b = (const float*)d_in[10];
  const float* bn2_g   = (const float*)d_in[11];
  const float* bn2_b   = (const float*)d_in[12];
  const float* fc_w    = (const float*)d_in[13];
  const float* fc_b    = (const float*)d_in[14];
  const float* hfc_w   = (const float*)d_in[15];
  const float* hfc_b   = (const float*)d_in[16];
  const float* w_ih    = (const float*)d_in[17];
  const float* w_hh    = (const float*)d_in[18];
  const float* b_ih    = (const float*)d_in[19];
  const float* b_hh    = (const float*)d_in[20];
  const float* go      = (const float*)d_in[21];
  const float* emb_w   = (const float*)d_in[22];
  const float* actor_w = (const float*)d_in[23];
  const float* actor_b = (const float*)d_in[24];
  const float* d1_w    = (const float*)d_in[25];
  const float* d1_b    = (const float*)d_in[26];
  const float* dc3_w   = (const float*)d_in[27];
  const float* dc3_b   = (const float*)d_in[28];
  const float* bnm2_g  = (const float*)d_in[29];
  const float* bnm2_b  = (const float*)d_in[30];
  const float* dc2_w   = (const float*)d_in[31];
  const float* dc2_b   = (const float*)d_in[32];
  const float* bnm1_g  = (const float*)d_in[33];
  const float* bnm1_b  = (const float*)d_in[34];
  const float* dc1_w   = (const float*)d_in[35];
  const float* dc1_b   = (const float*)d_in[36];

  float* out       = (float*)d_out;
  float* out_act   = out;                                        // [B,T,V]
  float* out_mask  = out_act  + (size_t)B_ * T_ * V_;            // [B,T,1,300,300]
  float* out_attn  = out_mask + (size_t)B_ * T_ * 300 * 300;     // [B,T,L,1]
  float* out_h     = out_attn + (size_t)B_ * T_ * L_;            // [B,DHID]
  float* out_c     = out_h    + (size_t)B_ * DHID;               // [B,DHID]

  // bump allocator on d_ws
  size_t off = 0;
  char* wsb = (char*)d_ws;
  auto aU = [&](size_t n) -> unsigned short* {
    unsigned short* p = (unsigned short*)(wsb + off);
    off = (off + n * 2 + 255) & ~(size_t)255; return p; };
  auto aF = [&](size_t n) -> float* {
    float* p = (float*)(wsb + off);
    off = (off + n * 4 + 255) & ~(size_t)255; return p; };

  // bf16 weights (N padded to mult of 16, K padded to mult of 32)
  unsigned short* wb_c1  = aU((size_t)256  * 512);
  unsigned short* wb_c2  = aU((size_t)64   * 256);
  unsigned short* wb_fc  = aU((size_t)2512 * FLAT);
  unsigned short* wb_hfc = aU((size_t)1024 * 1024);
  unsigned short* wb_ih  = aU((size_t)4096 * K32INP);
  unsigned short* wb_hh  = aU((size_t)4096 * 1024);
  unsigned short* wb_act = aU((size_t)128  * K32CONT);
  unsigned short* wb_emb = aU((size_t)1024 * 128);
  unsigned short* wb_d1  = aU((size_t)FLAT * K32CONT);
  // bf16 activation packs
  unsigned short* aFr  = aU((size_t)400 * 512);
  unsigned short* aC1  = aU((size_t)400 * 256);
  unsigned short* aFC  = aU((size_t)16 * FLAT);
  unsigned short* aH   = aU((size_t)16 * 1024);
  unsigned short* aINP = aU((size_t)16 * K32INP);
  unsigned short* aCON = aU((size_t)16 * K32CONT);
  unsigned short* aPRJ = aU((size_t)16 * 128);
  // fp32 scratch
  float* C1   = aF((size_t)392 * 256);
  float* C2   = aF((size_t)392 * 64);
  float* vis  = aF((size_t)B_ * DFRAME);
  float* hq   = aF((size_t)B_ * DHID);
  float* lgt  = aF((size_t)B_ * L_);
  float* scr  = aF((size_t)B_ * L_);
  float* wgt  = aF((size_t)B_ * DHID);
  float* gts  = aF((size_t)B_ * 4 * DHID);
  float* h    = aF((size_t)B_ * DHID);
  float* c    = aF((size_t)B_ * DHID);
  float* e    = aF((size_t)B_ * DEMB);
  float* prj  = aF((size_t)B_ * DEMB);
  float* d1o  = aF((size_t)B_ * FLAT);
  float* u3   = aF((size_t)B_ * 64 * 14 * 14);
  float* dc3o = aF((size_t)B_ * 32 * 28 * 28);
  float* u2b  = aF((size_t)B_ * 32 * 56 * 56);
  float* dc2o = aF((size_t)B_ * 16 * 112 * 112);
  float* dc1o = aF((size_t)B_ * 224 * 224);

  auto gemm = [&](const unsigned short* A, const unsigned short* Bw, float* C, int ldc,
                  const float* bias, const float* g, const float* bb,
                  int M, int N, int K32, int mode, int accum) {
    int tiles = ((M + 15) / 16) * ((N + 15) / 16);
    gemm_bf16_wmma<<<(tiles + 7) / 8, 256, 0, stream>>>(A, Bw, C, ldc, bias, g, bb,
                                                        M, N, K32, mode, accum);
  };
  auto pw = [&](const float* W, unsigned short* o, int N, int K, int K32, int Np) {
    pack_weight<<<gridFor((size_t)Np * K32), 256, 0, stream>>>(W, o, N, K, K32, Np);
  };

  // --- once per launch: weight conversion + state init ---
  pw(conv1_w, wb_c1,  256,  512,  512,  256);
  pw(conv2_w, wb_c2,  64,   256,  256,  64);
  pw(fc_w,    wb_fc,  2500, FLAT, FLAT, 2512);
  pw(hfc_w,   wb_hfc, 1024, 1024, 1024, 1024);
  pw(w_ih,    wb_ih,  4096, DINP, K32INP, 4096);
  pw(w_hh,    wb_hh,  4096, 1024, 1024, 4096);
  pw(actor_w, wb_act, 128,  DCONT, K32CONT, 128);
  pw(emb_w,   wb_emb, 1024, 128,  128,  1024);
  pw(d1_w,    wb_d1,  FLAT, DCONT, K32CONT, FLAT);
  init_state<<<gridFor(B_ * DHID), 256, 0, stream>>>(h0, c0, go, h, c, e);

  // --- time loop ---
  for (int t = 0; t < T_; ++t) {
    // vis_encode
    pack_frames<<<gridFor(400 * 512), 256, 0, stream>>>(frames, aFr, t);
    gemm(aFr, wb_c1, C1, 256, conv1_b, bn1_g, bn1_b, 392, 256, 512, 2, 0);
    pack_act<<<gridFor((size_t)400 * 256), 256, 0, stream>>>(C1, aC1, 392, 256, 400, 256);
    gemm(aC1, wb_c2, C2, 64, conv2_b, bn2_g, bn2_b, 392, 64, 256, 2, 0);
    pack_fc<<<gridFor(16 * FLAT), 256, 0, stream>>>(C2, aFC);
    gemm(aFC, wb_fc, vis, DFRAME, fc_b, nullptr, nullptr, B_, DFRAME, FLAT, 0, 0);
    // attention
    pack_act<<<gridFor((size_t)16 * 1024), 256, 0, stream>>>(h, aH, B_, 1024, 16, 1024);
    gemm(aH, wb_hfc, hq, DHID, hfc_b, nullptr, nullptr, B_, DHID, 1024, 0, 0);
    attn_logits<<<gridFor((size_t)B_ * L_ * 32) , 256, 0, stream>>>(enc, hq, lgt);
    softmax_attn<<<1, 256, 0, stream>>>(lgt, scr, out_attn, t);
    attn_weighted<<<gridFor(B_ * DHID), 256, 0, stream>>>(enc, scr, wgt);
    // LSTM
    pack_inp<<<gridFor((size_t)16 * K32INP), 256, 0, stream>>>(vis, wgt, e, gcn, aINP);
    gemm(aINP, wb_ih, gts, 4 * DHID, b_ih, nullptr, nullptr, B_, 4 * DHID, K32INP, 0, 0);
    gemm(aH,   wb_hh, gts, 4 * DHID, b_hh, nullptr, nullptr, B_, 4 * DHID, 1024, 0, 1);
    lstm_pw<<<gridFor(B_ * DHID), 256, 0, stream>>>(gts, h, c);
    // actor / action
    pack_cont<<<gridFor((size_t)16 * K32CONT), 256, 0, stream>>>(h, vis, wgt, e, gcn, aCON);
    gemm(aCON, wb_act, prj, DEMB, actor_b, nullptr, nullptr, B_, DEMB, K32CONT, 0, 0);
    pack_act<<<gridFor((size_t)16 * 128), 256, 0, stream>>>(prj, aPRJ, B_, 128, 16, 128);
    gemm(aPRJ, wb_emb, out_act + (size_t)t * V_, T_ * V_, nullptr, nullptr, nullptr,
         B_, V_, 128, 0, 0);
    argmax_embed<<<1, 256, 0, stream>>>(out_act + (size_t)t * V_, T_ * V_, emb_w, e);
    // mask decode
    gemm(aCON, wb_d1, d1o, FLAT, d1_b, nullptr, nullptr, B_, FLAT, K32CONT, 1, 0);
    up2_nearest<<<gridFor((size_t)B_ * 64 * 14 * 14), 256, 0, stream>>>(d1o, u3, 64, 7);
    deconv4s2<<<gridFor((size_t)B_ * 32 * 28 * 28), 256, 0, stream>>>(
        u3, dc3_w, dc3_b, bnm2_g, bnm2_b, dc3o, 64, 32, 14, 2);
    up2_nearest<<<gridFor((size_t)B_ * 32 * 56 * 56), 256, 0, stream>>>(dc3o, u2b, 32, 28);
    deconv4s2<<<gridFor((size_t)B_ * 16 * 112 * 112), 256, 0, stream>>>(
        u2b, dc2_w, dc2_b, bnm1_g, bnm1_b, dc2o, 32, 16, 56, 2);
    deconv4s2<<<gridFor((size_t)B_ * 224 * 224), 256, 0, stream>>>(
        dc2o, dc1_w, dc1_b, nullptr, nullptr, dc1o, 16, 1, 112, 0);
    resize_bilinear<<<gridFor((size_t)B_ * 300 * 300), 256, 0, stream>>>(dc1o, out_mask, t);
  }

  copy_hc<<<gridFor(B_ * DHID), 256, 0, stream>>>(h, c, out_h, out_c);
}
